// MultiHeadLatentAttention_35493609734914
// MI455X (gfx1250) — matrix operand order repaired and checked by reading, not compiled, not run on hardware
//
#include <hip/hip_runtime.h>

// ---------------------------------------------------------------------------
// CDNA5 (gfx1250) MLA forward.
//  - one-time f32->bf16 conversion of all GEMM operands (bandwidth pass)
//  - all GEMMs via v_wmma_f32_16x16x32_bf16, async global->LDS double-buffered
//  - fused attention: WMMA scores -> LDS softmax -> attn output -> WMMA context
// ---------------------------------------------------------------------------

typedef __bf16 v16bf __attribute__((ext_vector_type(16)));
typedef __bf16 v8bf  __attribute__((ext_vector_type(8)));
typedef float  v8f   __attribute__((ext_vector_type(8)));
typedef float  v4f   __attribute__((ext_vector_type(4)));
typedef int    v4i   __attribute__((ext_vector_type(4)));

#define WMMA_BF16(a, b, c)                                                    \
  __builtin_amdgcn_wmma_f32_16x16x32_bf16(false, (a), false, (b), (short)0,   \
                                          (c), false, false)

static __device__ __forceinline__ v16bf cat8(v8bf lo, v8bf hi) {
  return __builtin_shufflevector(lo, hi, 0, 1, 2, 3, 4, 5, 6, 7, 8, 9, 10, 11,
                                 12, 13, 14, 15);
}

// ---- async global->LDS (16B granularity), ASYNCcnt-tracked -----------------
#if defined(__HIP_DEVICE_COMPILE__) &&                                         \
    __has_builtin(__builtin_amdgcn_global_load_async_to_lds_b128)
#define HAS_ASYNC_LDS 1
#else
#define HAS_ASYNC_LDS 0
#endif

static __device__ __forceinline__ void async_cp16(const __bf16* g, __bf16* l) {
#if HAS_ASYNC_LDS
  __builtin_amdgcn_global_load_async_to_lds_b128(
      (__attribute__((address_space(1))) v4i*)(g),
      (__attribute__((address_space(3))) v4i*)(l), 0, 0);
#else
  *(v8bf*)l = *(const v8bf*)g;  // synchronous fallback
#endif
}

#if defined(__HIP_DEVICE_COMPILE__) &&                                         \
    __has_builtin(__builtin_amdgcn_s_wait_asynccnt)
#define WAIT_ASYNC(n) __builtin_amdgcn_s_wait_asynccnt(n)
#elif defined(__HIP_DEVICE_COMPILE__) && HAS_ASYNC_LDS
#define WAIT_ASYNC(n) asm volatile("s_wait_asynccnt " #n ::: "memory")
#else
#define WAIT_ASYNC(n) ((void)0)
#endif

// ---------------------------------------------------------------------------
// f32 -> bf16 conversion pass (8 elements / thread, 16B stores)
// ---------------------------------------------------------------------------
__global__ void f32_to_bf16_kernel(const float* __restrict__ src,
                                   __bf16* __restrict__ dst, int n8) {
  const int i = blockIdx.x * blockDim.x + threadIdx.x;
  if (i >= n8) return;
  const v4f* s = (const v4f*)src + (size_t)i * 2;
  v4f a = s[0], b = s[1];
  v8bf d;
  d[0] = (__bf16)a[0]; d[1] = (__bf16)a[1]; d[2] = (__bf16)a[2]; d[3] = (__bf16)a[3];
  d[4] = (__bf16)b[0]; d[5] = (__bf16)b[1]; d[6] = (__bf16)b[2]; d[7] = (__bf16)b[3];
  *((v8bf*)dst + i) = d;
}

// ---------------------------------------------------------------------------
// GEMM: C[M,N] = A[M,K] @ W[N,K]^T + bias.  A, W bf16; bias f32.
// Tile 128x128, BK=64, 256 threads = 8 waves, wave owns 32x64 (8 accumulators).
// Async global->LDS staging, double-buffered over the k-loop.
// ---------------------------------------------------------------------------
constexpr int BM = 128, BN = 128, BK = 64, LSTR = 72;  // LDS stride: 72 halves

__global__ __launch_bounds__(256) void gemm_bf16_wmma(
    const __bf16* __restrict__ A, const __bf16* __restrict__ W,
    const float* __restrict__ bias, float* __restrict__ Cf,
    __bf16* __restrict__ Cb, int M, int N, int K) {
  __shared__ __bf16 As[2][BM * LSTR];
  __shared__ __bf16 Ws[2][BN * LSTR];

  const int tid  = threadIdx.x;
  const int lane = tid & 31;
  const int w    = tid >> 5;
  const int hs   = lane >> 4;
  const int l16  = lane & 15;
  const int rowBase = blockIdx.y * BM;
  const int colBase = blockIdx.x * BN;
  const int wr = (w & 3) * 32;   // wave row offset (4 row-waves)
  const int wc = (w >> 2) * 64;  // wave col offset (2 col-waves)

  v8f acc[2][4] = {};
  const int nT = K / BK;

  auto prefetch = [&](int t, int buf) {
    const int k0 = t * BK;
#pragma unroll
    for (int i = 0; i < 4; ++i) {
      const int c  = tid + i * 256;  // 0..1023 16B-chunks per matrix
      const int r  = c >> 3;
      const int kc = (c & 7) * 8;
      async_cp16(A + (size_t)(rowBase + r) * K + k0 + kc,
                 &As[buf][r * LSTR + kc]);
      async_cp16(W + (size_t)(colBase + r) * K + k0 + kc,
                 &Ws[buf][r * LSTR + kc]);
    }
  };

  prefetch(0, 0);
  if (nT > 1) prefetch(1, 1);

  for (int t = 0; t < nT; ++t) {
    if (t + 1 < nT) { WAIT_ASYNC(8); } else { WAIT_ASYNC(0); }
    __syncthreads();  // tile t resident in LDS for every wave

    const __bf16* Ab = As[t & 1];
    const __bf16* Wb = Ws[t & 1];
#pragma unroll
    for (int ks = 0; ks < 2; ++ks) {  // BK=64 -> two k-steps of 32
      v16bf afr[2], bfr[4];
#pragma unroll
      for (int rt = 0; rt < 2; ++rt) {
        const __bf16* ap = &Ab[(wr + rt * 16 + l16) * LSTR + ks * 32 + hs * 8];
        afr[rt] = cat8(*(const v8bf*)ap, *(const v8bf*)(ap + 16));
      }
#pragma unroll
      for (int ct = 0; ct < 4; ++ct) {
        const __bf16* bp = &Wb[(wc + ct * 16 + l16) * LSTR + ks * 32 + hs * 8];
        bfr[ct] = cat8(*(const v8bf*)bp, *(const v8bf*)(bp + 16));
      }
#pragma unroll
      for (int rt = 0; rt < 2; ++rt)
#pragma unroll
        for (int ct = 0; ct < 4; ++ct)
          acc[rt][ct] = WMMA_BF16(afr[rt], bfr[ct], acc[rt][ct]);
    }
    __syncthreads();  // everyone done reading buffer (t&1)
    if (t + 2 < nT) prefetch(t + 2, t & 1);
  }

  // ---- epilogue ----
#pragma unroll
  for (int rt = 0; rt < 2; ++rt) {
#pragma unroll
    for (int ct = 0; ct < 4; ++ct) {
      const int col = colBase + wc + ct * 16 + l16;
      const float bv = bias[col];
#pragma unroll
      for (int r = 0; r < 8; ++r) {
        const int row = rowBase + wr + rt * 16 + hs * 8 + r;
        const float val = acc[rt][ct][r] + bv;
        const size_t off = (size_t)row * N + col;
        if (Cf) Cf[off] = val;
        if (Cb) Cb[off] = (__bf16)val;
      }
    }
  }
}

// ---------------------------------------------------------------------------
// RoPE + K assembly: k[b,s,h,0:64] = rope(k_rope_lin), k[b,s,h,64:128] =
// k_nope[b,s,h,64:128].
// ---------------------------------------------------------------------------
__global__ void rope_assemble_kernel(const float* __restrict__ kr,
                                     const __bf16* __restrict__ knope,
                                     __bf16* __restrict__ kfull) {
  const int idx = blockIdx.x * blockDim.x + threadIdx.x;  // B*S*H*32 = 2M
  if (idx >= 8 * 512 * 16 * 32) return;
  const int j   = idx & 31;
  const int h   = (idx >> 5) & 15;
  const int row = idx >> 9;    // b*512 + s
  const int t   = row & 511;   // sequence position
  const float invf = __expf(-0.28782313662425572f * (float)j);  // 10000^(-j/32)
  float sn, cs;
  __sincosf((float)t * invf, &sn, &cs);
  const float* kp = kr + (size_t)row * 1024 + h * 64;
  const float x1 = kp[j], x2 = kp[j + 32];
  __bf16* op = kfull + (size_t)row * 2048 + h * 128;
  op[j]      = (__bf16)(x1 * cs - x2 * sn);
  op[j + 32] = (__bf16)(x2 * cs + x1 * sn);
  const __bf16* np = knope + (size_t)row * 2048 + h * 128;
  op[64 + j] = np[64 + j];
  op[96 + j] = np[96 + j];
}

// ---------------------------------------------------------------------------
// Fused attention per (b, h, 32-query tile).  128 threads = 4 waves.
// ---------------------------------------------------------------------------
__global__ __launch_bounds__(128) void mla_attn_kernel(
    const __bf16* __restrict__ q, const __bf16* __restrict__ k,
    const __bf16* __restrict__ v, float* __restrict__ attn,
    __bf16* __restrict__ ctx) {
  __shared__ __bf16 Qs[32 * 136];   // [q_row][d], stride 136 halves
  __shared__ __bf16 KVs[9216];      // K: [key][d] s=136 ; V^T: [d][key] s=72
  __shared__ float  Ss[32 * 516];   // scores -> probabilities, stride 516
  __shared__ float  rowM[32], rowI[32];

  const int tid  = threadIdx.x;
  const int lane = tid & 31;
  const int w    = tid >> 5;
  const int hs   = lane >> 4;
  const int l16  = lane & 15;
  const int q0   = blockIdx.x * 32;
  const int h    = blockIdx.y;
  const int b    = blockIdx.z;
  const int rt   = w & 1;
  const float scale = 0.08838834764831845f;  // 1/sqrt(128)

  // ---- Q tile (32 x 128) : async into LDS ----
  {
    const int r = tid >> 2, part = tid & 3;
    const __bf16* gp =
        q + (size_t)(b * 512 + q0 + r) * 2048 + h * 128 + part * 32;
#pragma unroll
    for (int i = 0; i < 4; ++i)
      async_cp16(gp + i * 8, &Qs[r * 136 + part * 32 + i * 8]);
  }

  // ---- scores: stream K in 8 chunks of 64 keys ----
  for (int kc = 0; kc < 8; ++kc) {
    __syncthreads();  // previous chunk's compute done
    {
      const int r = tid >> 1, part = tid & 1;
      const __bf16* gp =
          k + (size_t)(b * 512 + kc * 64 + r) * 2048 + h * 128 + part * 64;
#pragma unroll
      for (int i = 0; i < 8; ++i)
        async_cp16(gp + i * 8, &KVs[r * 136 + part * 64 + i * 8]);
    }
    WAIT_ASYNC(0);
    __syncthreads();
#pragma unroll
    for (int cti = 0; cti < 2; ++cti) {
      const int ct = (w >> 1) + cti * 2;
      v8f acc = {};
#pragma unroll
      for (int ks = 0; ks < 4; ++ks) {  // dk=128 -> 4 k-steps of 32
        const __bf16* ap = &Qs[(rt * 16 + l16) * 136 + ks * 32 + hs * 8];
        v16bf af = cat8(*(const v8bf*)ap, *(const v8bf*)(ap + 16));
        const __bf16* bp = &KVs[(ct * 16 + l16) * 136 + ks * 32 + hs * 8];
        v16bf bf_ = cat8(*(const v8bf*)bp, *(const v8bf*)(bp + 16));
        acc = WMMA_BF16(af, bf_, acc);
      }
      const int colb = kc * 64 + ct * 16 + l16;
#pragma unroll
      for (int r = 0; r < 8; ++r)
        Ss[(rt * 16 + hs * 8 + r) * 516 + colb] = acc[r] * scale;
    }
  }
  __syncthreads();

  // ---- softmax reductions: 4 lanes per row ----
  {
    const int r = tid >> 2, sub = tid & 3;
    const float* sp = &Ss[r * 516 + sub * 128];
    float m = -3.4e38f;
    for (int c = 0; c < 128; ++c) m = fmaxf(m, sp[c]);
    m = fmaxf(m, __shfl_xor(m, 1, 32));
    m = fmaxf(m, __shfl_xor(m, 2, 32));
    float s = 0.0f;
    for (int c = 0; c < 128; ++c) s += __expf(sp[c] - m);
    s += __shfl_xor(s, 1, 32);
    s += __shfl_xor(s, 2, 32);
    if (sub == 0) { rowM[r] = m; rowI[r] = 1.0f / s; }
  }
  __syncthreads();

  // ---- normalize: write probabilities to global attn AND back into Ss ----
  for (int rr = 0; rr < 8; ++rr) {
    const int r = w * 8 + rr;
    const float m = rowM[r], inv = rowI[r];
    float* sp = &Ss[r * 516];
    float* gp = attn + ((size_t)(b * 16 + h) * 512 + q0 + r) * 512;
    for (int it = 0; it < 16; ++it) {
      const int c = it * 32 + lane;
      const float p = __expf(sp[c] - m) * inv;
      sp[c] = p;
      gp[c] = p;
    }
  }

  // ---- context = P @ V : stream V in 8 chunks, transposed into LDS ----
  v8f cacc[4] = {};
  for (int vc = 0; vc < 8; ++vc) {
    __syncthreads();
    {
      const int r = tid >> 1, part = tid & 1;  // r = key within chunk
      const __bf16* gp =
          v + (size_t)(b * 512 + vc * 64 + r) * 2048 + h * 128 + part * 64;
#pragma unroll
      for (int i = 0; i < 8; ++i) {
        v8bf d = *(const v8bf*)(gp + i * 8);
        const int dbase = part * 64 + i * 8;
#pragma unroll
        for (int jj = 0; jj < 8; ++jj)
          KVs[(dbase + jj) * 72 + r] = d[jj];  // transposed: [d][key]
      }
    }
    __syncthreads();
#pragma unroll
    for (int ks = 0; ks < 2; ++ks) {
      const int kofs = vc * 64 + ks * 32;
      const float* sp = &Ss[(rt * 16 + l16) * 516 + kofs + hs * 8];
      v16bf af{};
#pragma unroll
      for (int e = 0; e < 8; ++e) {
        af[e]     = (__bf16)sp[e];
        af[e + 8] = (__bf16)sp[e + 16];
      }
#pragma unroll
      for (int cti = 0; cti < 4; ++cti) {
        const int ct = (w >> 1) + cti * 2;
        const __bf16* bp = &KVs[(ct * 16 + l16) * 72 + ks * 32 + hs * 8];
        v16bf bf_ = cat8(*(const v8bf*)bp, *(const v8bf*)(bp + 16));
        cacc[cti] = WMMA_BF16(af, bf_, cacc[cti]);
      }
    }
  }

  // ---- write context (bf16, [b*512+row][h*128+d]) ----
#pragma unroll
  for (int cti = 0; cti < 4; ++cti) {
    const int ct = (w >> 1) + cti * 2;
    const int d  = ct * 16 + l16;
#pragma unroll
    for (int r = 0; r < 8; ++r) {
      const int row = q0 + rt * 16 + hs * 8 + r;
      ctx[(size_t)(b * 512 + row) * 2048 + h * 128 + d] = (__bf16)cacc[cti][r];
    }
  }
}

// ---------------------------------------------------------------------------
extern "C" void kernel_launch(void* const* d_in, const int* in_sizes, int n_in,
                              void* d_out, int out_size, void* d_ws,
                              size_t ws_size, hipStream_t stream) {
  (void)in_sizes; (void)n_in; (void)out_size; (void)ws_size;

  const float* q_in   = (const float*)d_in[0];
  const float* kv_in  = (const float*)d_in[1];
  const float* wq_w   = (const float*)d_in[2];
  const float* wq_b   = (const float*)d_in[3];
  const float* wdkv_w = (const float*)d_in[4];
  const float* wdkv_b = (const float*)d_in[5];
  const float* wupk_w = (const float*)d_in[6];
  const float* wupk_b = (const float*)d_in[7];
  const float* wupv_w = (const float*)d_in[8];
  const float* wupv_b = (const float*)d_in[9];
  const float* wkr_w  = (const float*)d_in[10];
  const float* wkr_b  = (const float*)d_in[11];
  const float* wo_w   = (const float*)d_in[12];
  const float* wo_b   = (const float*)d_in[13];

  float* out_f  = (float*)d_out;                // [4096, 2048]
  float* attn_f = out_f + (size_t)4096 * 2048;  // [8,16,512,512]

  char* ws = (char*)d_ws;
  auto carve_bf = [&](size_t n) { __bf16* p = (__bf16*)ws; ws += n * 2; return p; };
  auto carve_f  = [&](size_t n) { float*  p = (float*)ws;  ws += n * 4; return p; };

  __bf16* qin_bf  = carve_bf((size_t)4096 * 2048);
  __bf16* kvin_bf = carve_bf((size_t)4096 * 2048);
  __bf16* wq_bf   = carve_bf((size_t)2048 * 2048);
  __bf16* wdkv_bf = carve_bf((size_t)512 * 2048);
  __bf16* wupk_bf = carve_bf((size_t)2048 * 512);
  __bf16* wupv_bf = carve_bf((size_t)2048 * 512);
  __bf16* wkr_bf  = carve_bf((size_t)1024 * 2048);
  __bf16* wo_bf   = carve_bf((size_t)2048 * 2048);
  __bf16* q_bf    = carve_bf((size_t)4096 * 2048);
  __bf16* ckv_bf  = carve_bf((size_t)4096 * 512);
  __bf16* kn_bf   = carve_bf((size_t)4096 * 2048);
  __bf16* v_bf    = carve_bf((size_t)4096 * 2048);
  __bf16* kf_bf   = carve_bf((size_t)4096 * 2048);
  __bf16* ctx_bf  = carve_bf((size_t)4096 * 2048);
  float*  kr_f    = carve_f((size_t)4096 * 1024);

  // ---- conversion pass (bandwidth-bound, ~100 MB total) ----
  auto cvt = [&](const float* s, __bf16* d, size_t n) {
    const int n8 = (int)(n / 8);
    f32_to_bf16_kernel<<<(n8 + 255) / 256, 256, 0, stream>>>(s, d, n8);
  };
  cvt(q_in,   qin_bf,  (size_t)4096 * 2048);
  cvt(kv_in,  kvin_bf, (size_t)4096 * 2048);
  cvt(wq_w,   wq_bf,   (size_t)2048 * 2048);
  cvt(wdkv_w, wdkv_bf, (size_t)512 * 2048);
  cvt(wupk_w, wupk_bf, (size_t)2048 * 512);
  cvt(wupv_w, wupv_bf, (size_t)2048 * 512);
  cvt(wkr_w,  wkr_bf,  (size_t)1024 * 2048);
  cvt(wo_w,   wo_bf,   (size_t)2048 * 2048);

  const dim3 blk(256);
  // q = query_input @ wq^T + b                     (bf16 out)
  gemm_bf16_wmma<<<dim3(2048 / 128, 4096 / 128), blk, 0, stream>>>(
      qin_bf, wq_bf, wq_b, nullptr, q_bf, 4096, 2048, 2048);
  // c_kv = kv_input @ wdkv^T + b                   (bf16 out)
  gemm_bf16_wmma<<<dim3(512 / 128, 4096 / 128), blk, 0, stream>>>(
      kvin_bf, wdkv_bf, wdkv_b, nullptr, ckv_bf, 4096, 512, 2048);
  // k_nope = c_kv @ wupk^T + b                     (bf16 out)
  gemm_bf16_wmma<<<dim3(2048 / 128, 4096 / 128), blk, 0, stream>>>(
      ckv_bf, wupk_bf, wupk_b, nullptr, kn_bf, 4096, 2048, 512);
  // v = c_kv @ wupv^T + b                          (bf16 out)
  gemm_bf16_wmma<<<dim3(2048 / 128, 4096 / 128), blk, 0, stream>>>(
      ckv_bf, wupv_bf, wupv_b, nullptr, v_bf, 4096, 2048, 512);
  // k_rope_lin = kv_input @ wkr^T + b              (f32 out)
  gemm_bf16_wmma<<<dim3(1024 / 128, 4096 / 128), blk, 0, stream>>>(
      kvin_bf, wkr_bf, wkr_b, kr_f, nullptr, 4096, 1024, 2048);
  // rope + concat -> full K (bf16)
  rope_assemble_kernel<<<(8 * 512 * 16 * 32) / 256, 256, 0, stream>>>(
      kr_f, kn_bf, kf_bf);
  // fused attention: scores -> softmax -> attn out -> context
  mla_attn_kernel<<<dim3(512 / 32, 16, 8), dim3(128), 0, stream>>>(
      q_bf, kf_bf, v_bf, attn_f, ctx_bf);
  // out = context @ wo^T + b                       (f32 to d_out)
  gemm_bf16_wmma<<<dim3(2048 / 128, 4096 / 128), blk, 0, stream>>>(
      ctx_bf, wo_bf, wo_b, out_f, nullptr, 4096, 2048, 2048);
}
